// ShiftConvBlock_47485158425299
// MI455X (gfx1250) — compile-verified, matrix operand order checked by reading, not run on hardware
//
#include <hip/hip_runtime.h>
#include <hip/hip_bf16.h>

typedef __attribute__((ext_vector_type(16))) _Float16 v16h;
typedef __attribute__((ext_vector_type(8)))  float    v8f;
typedef __attribute__((ext_vector_type(4)))  unsigned u32x4;
typedef __attribute__((ext_vector_type(2)))  float    f32x2;

#define CIN   256
#define COUT  256
#define HDIM  56
#define WDIM  56
#define HW    3136          // 56*56
#define PITCH 264           // halves per LDS row (256 data + 8 pad) = 528 bytes
#define ROWS  128           // M rows per block
#define IMG   (CIN * HW)    // elements per image in x (and per image in out)

// Shifted-x gather: group g of channel c selects (dx,dy); y[h][w] = x[h-dy][w-dx] or 0.
__device__ __forceinline__ float load_shift(const float* __restrict__ x,
                                            int b, int c, int h, int w) {
    int g  = (c < 51) ? 0 : (c < 102) ? 1 : (c < 153) ? 2 : (c < 204) ? 3 : 4;
    int dx = (g == 1) ? 1 : (g == 2) ? -1 : 0;
    int dy = (g == 3) ? 1 : (g == 4) ? -1 : 0;
    int sh = h - dy;
    int sw = w - dx;
    if ((unsigned)sh < (unsigned)HDIM && (unsigned)sw < (unsigned)WDIM)
        return x[(size_t)b * IMG + (size_t)c * HW + sh * WDIM + sw];
    return 0.0f;
}

// Load one 16x32 f16 fragment (8 VGPRs) from an LDS row: two b128 reads.
__device__ __forceinline__ v16h ld_frag(const _Float16* p /* 16B aligned */) {
    struct Pack { u32x4 lo; u32x4 hi; } t;
    t.lo = *reinterpret_cast<const u32x4*>(p);
    t.hi = *reinterpret_cast<const u32x4*>(reinterpret_cast<const char*>(p) + 32);
    return __builtin_bit_cast(v16h, t);
}

extern "C" __global__ __launch_bounds__(256)
void shiftconv_wmma_kernel(const float* __restrict__ x,
                           const float* __restrict__ pw,
                           const float* __restrict__ gamma,
                           const float* __restrict__ beta,
                           const float* __restrict__ rmean,
                           const float* __restrict__ rvar,
                           float* __restrict__ out) {
    extern __shared__ char smem[];
    _Float16* sW  = reinterpret_cast<_Float16*>(smem);        // 256 * PITCH halves
    _Float16* sA  = sW + COUT * PITCH;                        // 128 * PITCH halves
    float*    sInv  = reinterpret_cast<float*>(sA + ROWS * PITCH);
    float*    sBias = sInv + COUT;

    const int tid = threadIdx.x;
    const int m0  = blockIdx.x * ROWS;

    // ---- BN affine precompute: inv = gamma*rsqrt(var+eps), bias = beta - mean*inv
    {
        int o = tid;  // blockDim.x == 256 == COUT
        float iv = gamma[o] * rsqrtf(rvar[o] + 1e-5f);
        sInv[o]  = iv;
        sBias[o] = beta[o] - rmean[o] * iv;
    }

    // ---- Stage weights (f32 global, L2-hot) -> f16 LDS, row-major [o][c], pitch 264
    // pair index p: o = p>>7, c = (p&127)*2 ; consecutive tid -> consecutive c (coalesced)
    for (int i = 0; i < 128; ++i) {
        int p = i * 256 + tid;
        int o = p >> 7;
        int c = (p & 127) << 1;
        f32x2 wv = *reinterpret_cast<const f32x2*>(pw + (size_t)o * CIN + c);
        union { _Float16 h[2]; unsigned u; } pk;
        pk.h[0] = (_Float16)wv.x;
        pk.h[1] = (_Float16)wv.y;
        *reinterpret_cast<unsigned*>(&sW[o * PITCH + c]) = pk.u;
    }

    // ---- Stage shifted A tile: 128 rows (m) x 256 cols (c) -> f16 LDS [m][c]
    // consecutive tid -> consecutive m for fixed c (coalesced global reads)
    for (int i = 0; i < 64; ++i) {
        int ml = tid & 127;
        int cp = i * 2 + (tid >> 7);
        int c  = cp << 1;
        int m  = m0 + ml;
        int b  = m / HW;
        int hw = m - b * HW;
        int h  = hw / WDIM;
        int w  = hw - h * WDIM;
        float v0 = load_shift(x, b, c,     h, w);
        float v1 = load_shift(x, b, c + 1, h, w);
        union { _Float16 hh[2]; unsigned u; } pk;
        pk.hh[0] = (_Float16)v0;
        pk.hh[1] = (_Float16)v1;
        *reinterpret_cast<unsigned*>(&sA[ml * PITCH + c]) = pk.u;
    }

    __syncthreads();

    // ---- Compute: wave `wv` owns m-tile rows [m0 + wv*16, +16)
    const int wave = tid >> 5;
    const int lane = tid & 31;
    const int ln   = lane & 15;
    const int upB  = (lane >> 4) ? 16 : 0;  // +16 bytes (K += 8) for upper lane half
    const int oAdd = (lane >> 4) ? 8  : 0;  // D frag: M += 8 for upper lane half

    // x fragments (B operand, 32x16 k x m), held for the whole o loop
    v16h bx[8];
    {
        const char* aBase = reinterpret_cast<const char*>(sA)
                          + (wave * 16 + ln) * (PITCH * 2) + upB;
        #pragma unroll
        for (int kc = 0; kc < 8; ++kc)
            bx[kc] = ld_frag(reinterpret_cast<const _Float16*>(aBase + kc * 64));
    }

    // output base for this lane's m (lanes 0..15 -> contiguous hw -> coalesced stores)
    const int m   = m0 + wave * 16 + ln;
    const int bI  = m / HW;
    const int hw  = m - bI * HW;
    float* outBase = out + (size_t)bI * IMG + hw;

    #pragma unroll 1
    for (int ot = 0; ot < 16; ++ot) {
        // weight fragments (A operand, 16x32 o x k) for this o-tile
        const char* wBase = reinterpret_cast<const char*>(sW)
                          + (ot * 16 + ln) * (PITCH * 2) + upB;
        v16h aw[8];
        #pragma unroll
        for (int kc = 0; kc < 8; ++kc)
            aw[kc] = ld_frag(reinterpret_cast<const _Float16*>(wBase + kc * 64));

        v8f acc = {};
        #pragma unroll
        for (int kc = 0; kc < 8; ++kc)
            acc = __builtin_amdgcn_wmma_f32_16x16x32_f16(
                      false, aw[kc], false, bx[kc],
                      (short)0, acc, false, false);

        // fused BN + ReLU, store D: VGPR v -> o = ot*16 + v + oAdd, N(lane) -> m
        #pragma unroll
        for (int v = 0; v < 8; ++v) {
            int o = ot * 16 + v + oAdd;
            float r = acc[v] * sInv[o] + sBias[o];
            r = r > 0.0f ? r : 0.0f;
            outBase[(size_t)o * HW] = r;
        }
    }
}

extern "C" void kernel_launch(void* const* d_in, const int* in_sizes, int n_in,
                              void* d_out, int out_size, void* d_ws, size_t ws_size,
                              hipStream_t stream) {
    const float* x     = (const float*)d_in[0];
    const float* pw    = (const float*)d_in[1];
    const float* gamma = (const float*)d_in[2];
    const float* beta  = (const float*)d_in[3];
    const float* rmean = (const float*)d_in[4];
    const float* rvar  = (const float*)d_in[5];
    float* out = (float*)d_out;

    const int M = 32 * HW;                 // 100352
    const int nblocks = M / ROWS;          // 784, exact
    const size_t shmem = (size_t)(COUT * PITCH + ROWS * PITCH) * sizeof(_Float16)
                       + 2 * COUT * sizeof(float);   // 204800 bytes < 320KB WGP LDS

    shiftconv_wmma_kernel<<<nblocks, 256, shmem, stream>>>(
        x, pw, gamma, beta, rmean, rvar, out);
}
